// GraphSimCLR_31774168056043
// MI455X (gfx1250) — compile-verified, matrix operand order checked by reading, not run on hardware
//
#include <hip/hip_runtime.h>

typedef __attribute__((ext_vector_type(16))) _Float16 v16h;
typedef __attribute__((ext_vector_type(8)))  _Float16 v8h;
typedef __attribute__((ext_vector_type(8)))  float    v8f;

#define NN   100000
#define EE   300000
#define HID  256
#define BN_EPS 1e-5f

// ---------------- small utility kernels ----------------

// layer-1 GCN linear: h[i,c] = x[i] * W[0,c]   (Kin == 1)
__global__ void outer_l1_kernel(const float* __restrict__ x,
                                const float* __restrict__ w,
                                float* __restrict__ h) {
    int i = blockIdx.x;
    int c = threadIdx.x;
    h[(long)i * HID + c] = x[i] * w[c];
}

__global__ void deg_init_kernel(float* __restrict__ deg, int n) {
    int i = blockIdx.x * blockDim.x + threadIdx.x;
    if (i < n) deg[i] = 1.0f;   // self-loop contributes 1 to every node's degree
}

__global__ void deg_edge_kernel(const int* __restrict__ dst,
                                float* __restrict__ deg, int e) {
    int i = blockIdx.x * blockDim.x + threadIdx.x;
    if (i < e) atomicAdd(&deg[dst[i]], 1.0f);
}

// in-place deg -> dis = rsqrt(deg)
__global__ void dis_kernel(float* __restrict__ deg, int n) {
    int i = blockIdx.x * blockDim.x + threadIdx.x;
    if (i < n) {
        float d = deg[i];
        deg[i] = d > 0.0f ? rsqrtf(d) : 0.0f;
    }
}

// one block per edge: acc[dst] += h[src] * dis[src]*dis[dst]
__global__ void scatter_kernel(const float* __restrict__ h,
                               const int* __restrict__ src,
                               const int* __restrict__ dst,
                               const float* __restrict__ dis,
                               float* __restrict__ acc) {
    int e = blockIdx.x;
    int s = src[e];
    int d = dst[e];
    float w = dis[s] * dis[d];
    int c = threadIdx.x;
    atomicAdd(&acc[(long)d * HID + c], h[(long)s * HID + c] * w);
}

// out = relu(acc + h*dis^2 + bias)
__global__ void finish_relu_kernel(const float* __restrict__ acc,
                                   const float* __restrict__ h,
                                   const float* __restrict__ dis,
                                   const float* __restrict__ bias,
                                   float* __restrict__ out) {
    int i = blockIdx.x;
    int c = threadIdx.x;
    long idx = (long)i * HID + c;
    float di = dis[i];
    float v = acc[idx] + h[idx] * di * di + bias[c];
    out[idx] = fmaxf(v, 0.0f);
}

// per-channel sum / sumsq, 64 rows per block, coalesced row reads
__global__ void bn_stats_kernel(const float* __restrict__ z,
                                float* __restrict__ sums,
                                float* __restrict__ sqs, int n) {
    int c = threadIdx.x;
    int r0 = blockIdx.x * 64;
    int rend = r0 + 64;
    if (rend > n) rend = n;
    float s = 0.0f, q = 0.0f;
    for (int r = r0; r < rend; ++r) {
        float v = z[(long)r * HID + c];
        s += v;
        q += v * v;
    }
    atomicAdd(&sums[c], s);
    atomicAdd(&sqs[c], q);
}

__global__ void bn_apply_kernel(const float* __restrict__ z,
                                const float* __restrict__ sums,
                                const float* __restrict__ sqs,
                                const float* __restrict__ gamma,
                                const float* __restrict__ beta,
                                float* __restrict__ out) {
    int i = blockIdx.x;
    int c = threadIdx.x;
    const float invn = 1.0f / (float)NN;
    float mu  = sums[c] * invn;
    float var = sqs[c] * invn - mu * mu;
    long idx = (long)i * HID + c;
    out[idx] = gamma[c] * (z[idx] - mu) * rsqrtf(var + BN_EPS) + beta[c];
}

// activation fp32 -> fp16 (vectorized 4-wide: b128 load, b64 store)
__global__ void act_to_f16_kernel(const float* __restrict__ in,
                                  _Float16* __restrict__ out, long n4) {
    long i = (long)blockIdx.x * blockDim.x + threadIdx.x;
    if (i < n4) {
        const float4 v = ((const float4*)in)[i];
        v8h* o = (v8h*)out;   // write 4 halves = 8B via half of a v8h slot
        _Float16* p = out + i * 4;
        p[0] = (_Float16)v.x; p[1] = (_Float16)v.y;
        p[2] = (_Float16)v.z; p[3] = (_Float16)v.w;
        (void)o;
    }
}

// ---------------- weight packing into WMMA B-fragment order ----------------
// For each (k-tile kt, col-tile ct), lane L needs 16 halves:
//   col = ct*16 + (L&15), kh = (L>>4)*8
//   e in 0..7  -> W[kt*32 + kh + e       , col]
//   e in 8..15 -> W[kt*32 + 16 + kh + e-8, col]
// Packed layout: P[((kt*colTiles + ct)*32 + L)*16 + e]  (32 B contiguous per lane)
__global__ void pack_b_kernel(const float* __restrict__ W,
                              _Float16* __restrict__ P,
                              int Kout, int n) {
    int tid = blockIdx.x * blockDim.x + threadIdx.x;
    if (tid >= n) return;
    int e    = tid & 15;
    int lane = (tid >> 4) & 31;
    int tile = tid >> 9;
    int colTiles = Kout >> 4;
    int ct = tile % colTiles;
    int kt = tile / colTiles;
    int col = ct * 16 + (lane & 15);
    int kh  = (lane >> 4) * 8;
    int k   = kt * 32 + (e < 8 ? (kh + e) : (16 + kh + (e - 8)));
    P[tid] = (_Float16)W[(long)k * Kout + col];
}

// ---------------- WMMA GEMM ----------------
// C[N, Kout] = act(A[N, KIN](f16) @ W[KIN, Kout](packed f16) + bias)
// block = 256 threads = 8 waves; each wave owns a 16x32 output tile
// (one A fragment drives two WMMAs per k-step). K fully unrolled at compile time.
// grid = (N/16, Kout/256). All dims exact -> EXEC all-ones around WMMA.
template <int KIN>
__global__ __launch_bounds__(256)
void gemm_wmma_kernel(const _Float16* __restrict__ A,
                      const _Float16* __restrict__ Bp,
                      const float* __restrict__ bias,
                      float* __restrict__ C,
                      int Kout, int has_bias, int relu) {
    int lane = threadIdx.x & 31;
    int wave = threadIdx.x >> 5;
    int row0 = blockIdx.x * 16;
    int ct0  = (blockIdx.y * 8 + wave) * 2;   // two 16-col tiles per wave
    int colTiles = Kout >> 4;

    int rsub = lane & 15;
    int khs  = lane >> 4;          // 0 or 1: selects K sub-group (halves of 8)
    int col0 = ct0 * 16 + rsub;
    int col1 = col0 + 16;

    float bv0 = has_bias ? bias[col0] : 0.0f;
    float bv1 = has_bias ? bias[col1] : 0.0f;
    v8f acc0, acc1;
#pragma unroll
    for (int j = 0; j < 8; ++j) { acc0[j] = bv0; acc1[j] = bv1; }

    // A row base for this lane; v8h-granular (8 halves = 16 B, aligned:
    // row stride = KIN*2 bytes = 512 or 1024)
    const v8h*  ap  = (const v8h*)(A + (long)(row0 + rsub) * KIN);
    const v16h* bfr = (const v16h*)Bp;       // fragment-granular indexing

#pragma unroll
    for (int kt = 0; kt < KIN / 32; ++kt) {
        // groups of 8 halves: [kt*32 + khs*8 .. +7] and [kt*32 + 16 + khs*8 .. +7]
        v8h g0 = ap[kt * 4 + khs];
        v8h g1 = ap[kt * 4 + 2 + khs];
        v16h a = __builtin_shufflevector(g0, g1,
                                         0, 1, 2, 3, 4, 5, 6, 7,
                                         8, 9, 10, 11, 12, 13, 14, 15);
        v16h b0 = bfr[((long)kt * colTiles + ct0)     * 32 + lane];
        v16h b1 = bfr[((long)kt * colTiles + ct0 + 1) * 32 + lane];
        acc0 = __builtin_amdgcn_wmma_f32_16x16x32_f16(
            false, a, false, b0, (short)0, acc0, false, false);
        acc1 = __builtin_amdgcn_wmma_f32_16x16x32_f16(
            false, a, false, b1, (short)0, acc1, false, false);
    }

    // D layout: VGPR j -> row (row0 + j + 8*(lane>>4)), col = tile base + (lane&15)
    int rhi = khs * 8;
#pragma unroll
    for (int j = 0; j < 8; ++j) {
        float v0 = acc0[j];
        float v1 = acc1[j];
        if (relu) { v0 = fmaxf(v0, 0.0f); v1 = fmaxf(v1, 0.0f); }
        long r = (long)(row0 + j + rhi);
        C[r * Kout + col0] = v0;
        C[r * Kout + col1] = v1;
    }
}

// ---------------- launcher ----------------

extern "C" void kernel_launch(void* const* d_in, const int* in_sizes, int n_in,
                              void* d_out, int out_size, void* d_ws, size_t ws_size,
                              hipStream_t stream) {
    (void)in_sizes; (void)n_in; (void)out_size; (void)ws_size;

    const float* x1  = (const float*)d_in[0];
    const int*   ei1 = (const int*)  d_in[1];
    const float* x2  = (const float*)d_in[2];
    const int*   ei2 = (const int*)  d_in[3];
    const float* Wg1 = (const float*)d_in[4];
    const float* bg1 = (const float*)d_in[5];
    const float* Wg2 = (const float*)d_in[6];
    const float* bg2 = (const float*)d_in[7];
    const float* Wg3 = (const float*)d_in[8];
    const float* bg3 = (const float*)d_in[9];
    const float* gamma1 = (const float*)d_in[10];
    const float* beta1  = (const float*)d_in[11];
    const float* gamma2 = (const float*)d_in[12];
    const float* beta2  = (const float*)d_in[13];
    const float* gamma3 = (const float*)d_in[14];
    const float* beta3  = (const float*)d_in[15];
    const float* Wp1 = (const float*)d_in[16];
    const float* bp1 = (const float*)d_in[17];
    const float* Wp2 = (const float*)d_in[18];
    const float* bp2 = (const float*)d_in[19];
    const float* Wp3 = (const float*)d_in[20];
    const float* bp3 = (const float*)d_in[21];

    float* out = (float*)d_out;
    const long NS = (long)NN * HID;          // elements per [N,256] tensor
    const size_t SZ = (size_t)NS * 4;        // bytes

    char* ws = (char*)d_ws;
    float*    bufH   = (float*)(ws);                 // [N,256] linear output / proj hidden2
    float*    bufAcc = (float*)(ws + SZ);            // [N,256] scatter acc / f16 act staging
    float*    buf512 = bufH;                         // union of bufH+bufAcc = [N,512] f32
    float*    bufZ   = (float*)(ws + 2 * SZ);        // [N,256] activation / f16 staging
    float*    deg    = (float*)(ws + 3 * SZ);        // [N] degree -> dis (in place)
    float*    sums   = (float*)(ws + 3 * SZ + (size_t)NN * 4);       // [256]
    float*    sqs    = sums + HID;                                    // [256]
    // packed B-fragment weights (32-byte aligned given ws alignment)
    _Float16* Wg2p = (_Float16*)((char*)(sqs + HID));                // 256x256
    _Float16* Wg3p = Wg2p + 256 * 256;
    _Float16* Wp1p = Wg3p + 256 * 256;                               // 256x512
    _Float16* Wp2p = Wp1p + 256 * 512;                               // 512x256
    _Float16* Wp3p = Wp2p + 512 * 256;                               // 256x256

    // ---- one-time weight packing into WMMA fragment order ----
    {
        int n = 256 * 256;
        pack_b_kernel<<<(n + 255) / 256, 256, 0, stream>>>(Wg2, Wg2p, 256, n);
        pack_b_kernel<<<(n + 255) / 256, 256, 0, stream>>>(Wg3, Wg3p, 256, n);
        pack_b_kernel<<<(n + 255) / 256, 256, 0, stream>>>(Wp3, Wp3p, 256, n);
        int m = 256 * 512;
        pack_b_kernel<<<(m + 255) / 256, 256, 0, stream>>>(Wp1, Wp1p, 512, m);
        pack_b_kernel<<<(m + 255) / 256, 256, 0, stream>>>(Wp2, Wp2p, 256, m);
    }

    const int  gridN   = NN / 16;            // 6250 row tiles
    const int  statGrd = (NN + 63) / 64;
    const long n4_256  = NS / 4;             // float4 count for [N,256]
    const long n4_512  = NS / 2;             // float4 count for [N,512]
    const int  cvtG256 = (int)((n4_256 + 255) / 256);
    const int  cvtG512 = (int)((n4_512 + 255) / 256);

    for (int view = 0; view < 2; ++view) {
        const float* x   = view == 0 ? x1 : x2;
        const int*   ei  = view == 0 ? ei1 : ei2;
        const int*   src = ei;            // ei[0,:]
        const int*   dst = ei + EE;       // ei[1,:]
        float* zout = out + (view == 0 ? 0 : NS);           // z slice in d_out
        float* pout = out + (view == 0 ? 2 * NS : 3 * NS);  // p slice in d_out

        // ---- degrees -> dis ----
        deg_init_kernel<<<(NN + 255) / 256, 256, 0, stream>>>(deg, NN);
        deg_edge_kernel<<<(EE + 255) / 256, 256, 0, stream>>>(dst, deg, EE);
        dis_kernel<<<(NN + 255) / 256, 256, 0, stream>>>(deg, NN);
        const float* dis = deg;

        // ---- 3 GCN layers ----
        for (int layer = 0; layer < 3; ++layer) {
            const float* bias  = layer == 0 ? bg1 : (layer == 1 ? bg2 : bg3);
            const float* gam   = layer == 0 ? gamma1 : (layer == 1 ? gamma2 : gamma3);
            const float* bet   = layer == 0 ? beta1 : (layer == 1 ? beta2 : beta3);
            float* zdst = (layer == 2) ? zout : bufZ;

            // linear transform -> bufH
            if (layer == 0) {
                outer_l1_kernel<<<NN, HID, 0, stream>>>(x, Wg1, bufH);
            } else {
                // stage activations as f16 in bufAcc (dead until the memset below)
                _Float16* Ah = (_Float16*)bufAcc;
                act_to_f16_kernel<<<cvtG256, 256, 0, stream>>>(bufZ, Ah, n4_256);
                const _Float16* Wp = (layer == 1) ? Wg2p : Wg3p;
                dim3 g(gridN, HID / 256);   // = (6250, 1)
                gemm_wmma_kernel<256><<<g, 256, 0, stream>>>(Ah, Wp, nullptr, bufH,
                                                             HID, /*bias*/0, /*relu*/0);
            }

            // scatter-add with symmetric normalization
            hipMemsetAsync(bufAcc, 0, SZ, stream);
            scatter_kernel<<<EE, HID, 0, stream>>>(bufH, src, dst, dis, bufAcc);

            // + self-loop + bias, ReLU -> zdst
            finish_relu_kernel<<<NN, HID, 0, stream>>>(bufAcc, bufH, dis, bias, zdst);

            // BatchNorm (training stats over node dim)
            hipMemsetAsync(sums, 0, 2 * HID * sizeof(float), stream);
            bn_stats_kernel<<<statGrd, HID, 0, stream>>>(zdst, sums, sqs, NN);
            bn_apply_kernel<<<NN, HID, 0, stream>>>(zdst, sums, sqs, gam, bet, zdst);
        }

        // ---- projector: relu(z@Wp1+bp1) -> relu(@Wp2+bp2) -> @Wp3+bp3 ----
        {
            // proj1: A = zout(f16 in bufZ, now free), out = buf512 (bufH+bufAcc)
            _Float16* Zh = (_Float16*)bufZ;
            act_to_f16_kernel<<<cvtG256, 256, 0, stream>>>(zout, Zh, n4_256);
            dim3 g1(gridN, 512 / 256);      // (6250, 2)
            gemm_wmma_kernel<256><<<g1, 256, 0, stream>>>(Zh, Wp1p, bp1, buf512,
                                                          512, /*bias*/1, /*relu*/1);
            // proj2: A = buf512 (f16 staged in bufZ; f32 source dead after convert),
            //        out = bufH (overwrites first half of buf512, no longer read)
            _Float16* H1h = (_Float16*)bufZ;
            act_to_f16_kernel<<<cvtG512, 256, 0, stream>>>(buf512, H1h, n4_512);
            dim3 g2(gridN, 256 / 256);      // (6250, 1)
            gemm_wmma_kernel<512><<<g2, 256, 0, stream>>>(H1h, Wp2p, bp2, bufH,
                                                          256, /*bias*/1, /*relu*/1);
            // proj3: A = bufH (f16 staged in bufAcc), out = pout
            _Float16* H2h = (_Float16*)bufAcc;
            act_to_f16_kernel<<<cvtG256, 256, 0, stream>>>(bufH, H2h, n4_256);
            dim3 g3(gridN, 256 / 256);      // (6250, 1)
            gemm_wmma_kernel<256><<<g3, 256, 0, stream>>>(H2h, Wp3p, bp3, pout,
                                                          256, /*bias*/1, /*relu*/0);
        }
    }
}
